// Matcher_7739531067855
// MI455X (gfx1250) — compile-verified, matrix operand order checked by reading, not run on hardware
//
#include <hip/hip_runtime.h>
#include <hip/hip_bf16.h>
#include <cmath>

typedef __attribute__((ext_vector_type(16))) _Float16 v16h;
typedef __attribute__((ext_vector_type(8)))  _Float16 v8h;
typedef __attribute__((ext_vector_type(8)))  float    v8f;

// ---------------------------------------------------------------------------
// Fragment layout for V_WMMA_F32_16X16X32_F16 (CDNA5 ISA 7.12.2):
//   A (16x32 f16): lane = m + 16*((k>>3)&1), slot = (k&7) | ((k>>4)<<3)
//   B (32x16 f16): same with n in place of m.
//   C/D: vgpr r -> m = r + 8*(lane>>4), n = lane&15.
// For k in [kb, kb+8) with kb % 8 == 0: lane fixed, slots contiguous
// -> 8 halves = one 16B LDS store per thread.
// ---------------------------------------------------------------------------

#define BM 128
#define BN 64
// Requires: M % 16 == 0 (epilogue guarded), N % BN == 0, K % 32 == 0.
// All GEMMs in this workload satisfy this.
__global__ __launch_bounds__(256)
void gemm_wmma(const float* __restrict__ A, const float* __restrict__ B,
               float* __restrict__ C, int M, int N, int K,
               int as_m, int as_k, int bs_k, int bs_n, int cs_m, int cs_n,
               const float* __restrict__ bias, const float* __restrict__ res,
               float scale, int relu)
{
    __shared__ __align__(32) _Float16 sA[2][BM * 32];   // 2 x 8 KB, fragment order
    __shared__ __align__(32) _Float16 sB[2][32 * BN];   // 2 x 4 KB, fragment order
    const int tid  = threadIdx.x;
    const int lane = tid & 31;
    const int wave = tid >> 5;                 // 0..7 -> 16-row slab of BM
    const int m0 = blockIdx.y * BM;
    const int n0 = blockIdx.x * BN;

    // staging: thread -> (row srow, 8 consecutive k at skb)
    const int srow = tid >> 2;                 // 0..63
    const int skb  = (tid & 3) * 8;            // 0,8,16,24
    const int lane_half = ((skb >> 3) & 1) << 4;
    const int slot0     = (skb >> 4) << 3;
    // LDS fragment offsets (in halves); 16B aligned
    const int stoff = ((srow >> 4) << 9) + (((srow & 15) + lane_half) << 4) + slot0;

    // 32-bit global element offsets (all buffers < 2^31 elems)
    const int aoff0 = (m0 + srow) * as_m + skb * as_k;
    const int aoff1 = (m0 + srow + 64) * as_m + skb * as_k;
    const int boff  = skb * bs_k + (n0 + srow) * bs_n;
    const int astep = 32 * as_k;               // advance per K tile
    const int bstep = 32 * bs_k;

    // fetch 8 elems (stride in elements) -> f[8]
    auto fetch8 = [&](const float* __restrict__ P, int off, int stride, float f[8]) {
        if (stride == 1) {
            float4 p0 = *(const float4*)(P + off);
            float4 p1 = *(const float4*)(P + off + 4);
            f[0] = p0.x; f[1] = p0.y; f[2] = p0.z; f[3] = p0.w;
            f[4] = p1.x; f[5] = p1.y; f[6] = p1.z; f[7] = p1.w;
        } else {
            #pragma unroll
            for (int j = 0; j < 8; ++j) f[j] = P[off + j * stride];
        }
    };
    auto pack8 = [&](const float f[8]) {
        v8h h;
        #pragma unroll
        for (int j = 0; j < 8; ++j) h[j] = (_Float16)f[j];
        return h;
    };

    v8f acc[4] = {{}, {}, {}, {}};
    float fA0[8], fA1[8], fB[8];

    // prologue: stage tile 0 into buffer 0
    fetch8(A, aoff0, as_k, fA0);
    fetch8(A, aoff1, as_k, fA1);
    fetch8(B, boff,  bs_k, fB);
    *(v8h*)&sA[0][stoff]        = pack8(fA0);
    *(v8h*)&sA[0][stoff + 2048] = pack8(fA1);
    *(v8h*)&sB[0][stoff]        = pack8(fB);
    __syncthreads();

    for (int k0 = 0; k0 < K; k0 += 32) {
        const int cur = (k0 >> 5) & 1;
        const int t = k0 >> 5;
        // stage next tile into the other buffer while computing this one
        if (k0 + 32 < K) {
            if (k0 + 64 < K) {
                __builtin_prefetch(A + aoff0 + (t + 2) * astep, 0, 1);
                __builtin_prefetch(B + boff  + (t + 2) * bstep, 0, 1);
            }
            fetch8(A, aoff0 + (t + 1) * astep, as_k, fA0);
            fetch8(A, aoff1 + (t + 1) * astep, as_k, fA1);
            fetch8(B, boff  + (t + 1) * bstep, bs_k, fB);
            *(v8h*)&sA[cur ^ 1][stoff]        = pack8(fA0);
            *(v8h*)&sA[cur ^ 1][stoff + 2048] = pack8(fA1);
            *(v8h*)&sB[cur ^ 1][stoff]        = pack8(fB);
        }
        // one A fragment, four distinct B fragments -> 4 back-to-back WMMAs
        v16h a  = *(const v16h*)&sA[cur][(wave << 9) + (lane << 4)];
        v16h b0 = *(const v16h*)&sB[cur][(0 << 9) + (lane << 4)];
        v16h b1 = *(const v16h*)&sB[cur][(1 << 9) + (lane << 4)];
        v16h b2 = *(const v16h*)&sB[cur][(2 << 9) + (lane << 4)];
        v16h b3 = *(const v16h*)&sB[cur][(3 << 9) + (lane << 4)];
        acc[0] = __builtin_amdgcn_wmma_f32_16x16x32_f16(false, a, false, b0, (short)0, acc[0], false, false);
        acc[1] = __builtin_amdgcn_wmma_f32_16x16x32_f16(false, a, false, b1, (short)0, acc[1], false, false);
        acc[2] = __builtin_amdgcn_wmma_f32_16x16x32_f16(false, a, false, b2, (short)0, acc[2], false, false);
        acc[3] = __builtin_amdgcn_wmma_f32_16x16x32_f16(false, a, false, b3, (short)0, acc[3], false, false);
        __syncthreads();   // next tile's stores done; this tile's reads done
    }

    // ---- epilogue: C layout (vgpr r, lane l) -> m = r + 8*(l>>4), n = l&15
    const int mb = m0 + wave * 16 + ((lane >> 4) << 3);
    const int nb = n0 + (lane & 15);
    #pragma unroll
    for (int r = 0; r < 8; ++r) {
        const int m = mb + r;
        if (m >= M) continue;
        const float bv = bias ? bias[m] : 0.f;
        #pragma unroll
        for (int j = 0; j < 4; ++j) {
            const int n = nb + j * 16;
            float v = acc[j][r] * scale + bv;
            const int ci = m * cs_m + n * cs_n;
            if (res) v += res[ci];
            if (relu) v = v > 0.f ? v : 0.f;
            C[ci] = v;
        }
    }
}

// Small 1x1 conv (keypoint encoder only, K<=128)
__global__ void conv_naive(const float* __restrict__ X, const float* __restrict__ W,
                           const float* __restrict__ bias, const float* __restrict__ res,
                           float* __restrict__ Y, int O, int Cc, int Nlen, int relu)
{
    int n = blockIdx.x * blockDim.x + threadIdx.x;
    int o = blockIdx.y;
    if (n >= Nlen) return;
    float acc = bias ? bias[o] : 0.f;
    for (int c = 0; c < Cc; ++c) acc += W[o * Cc + c] * X[c * Nlen + n];
    if (res) acc += res[o * Nlen + n];
    if (relu) acc = acc > 0.f ? acc : 0.f;
    Y[o * Nlen + n] = acc;
}

// InstanceNorm1d (affine=False) over length, optional ReLU; one block per channel
__global__ __launch_bounds__(256)
void inorm_kernel(float* __restrict__ X, int Nlen, int relu)
{
    __shared__ float red[256];
    float* row = X + (size_t)blockIdx.x * Nlen;
    float s = 0.f;
    for (int i = threadIdx.x; i < Nlen; i += 256) s += row[i];
    red[threadIdx.x] = s; __syncthreads();
    for (int o = 128; o > 0; o >>= 1) {
        if (threadIdx.x < o) red[threadIdx.x] += red[threadIdx.x + o];
        __syncthreads();
    }
    float mean = red[0] / Nlen; __syncthreads();
    float s2 = 0.f;
    for (int i = threadIdx.x; i < Nlen; i += 256) { float d = row[i] - mean; s2 += d * d; }
    red[threadIdx.x] = s2; __syncthreads();
    for (int o = 128; o > 0; o >>= 1) {
        if (threadIdx.x < o) red[threadIdx.x] += red[threadIdx.x + o];
        __syncthreads();
    }
    float inv = rsqrtf(red[0] / Nlen + 1e-5f);
    for (int i = threadIdx.x; i < Nlen; i += 256) {
        float v = (row[i] - mean) * inv;
        if (relu) v = v > 0.f ? v : 0.f;
        row[i] = v;
    }
}

// Row softmax, in place
__global__ __launch_bounds__(256)
void softmax_rows(float* __restrict__ S, int cols)
{
    __shared__ float red[256];
    float* row = S + (size_t)blockIdx.x * cols;
    float mx = -1e30f;
    for (int i = threadIdx.x; i < cols; i += 256) mx = fmaxf(mx, row[i]);
    red[threadIdx.x] = mx; __syncthreads();
    for (int o = 128; o > 0; o >>= 1) {
        if (threadIdx.x < o) red[threadIdx.x] = fmaxf(red[threadIdx.x], red[threadIdx.x + o]);
        __syncthreads();
    }
    mx = red[0]; __syncthreads();
    float s = 0.f;
    for (int i = threadIdx.x; i < cols; i += 256) { float e = __expf(row[i] - mx); row[i] = e; s += e; }
    red[threadIdx.x] = s; __syncthreads();
    for (int o = 128; o > 0; o >>= 1) {
        if (threadIdx.x < o) red[threadIdx.x] += red[threadIdx.x + o];
        __syncthreads();
    }
    float inv = 1.f / red[0];
    for (int i = threadIdx.x; i < cols; i += 256) row[i] *= inv;
}

// [kpts^T ; scores] -> kin[4 x N]
__global__ void build_kin(const float* __restrict__ kpts, const float* __restrict__ sc,
                          float* __restrict__ kin, int Nlen)
{
    int n = blockIdx.x * blockDim.x + threadIdx.x;
    if (n >= Nlen) return;
    kin[0 * Nlen + n] = kpts[n * 3 + 0];
    kin[1 * Nlen + n] = kpts[n * 3 + 1];
    kin[2 * Nlen + n] = kpts[n * 3 + 2];
    kin[3 * Nlen + n] = sc[n];
}

// Augment scores with dustbin row/col (alpha) -> couplings Z [R x R]
__global__ void build_couplings(const float* __restrict__ scores, const float* __restrict__ alpha,
                                float* __restrict__ Z, int R)
{
    int j = blockIdx.x * blockDim.x + threadIdx.x;
    int i = blockIdx.y;
    if (j >= R) return;
    float v = (i < R - 1 && j < R - 1) ? scores[i * (R - 1) + j] : alpha[0];
    Z[i * R + j] = v;
}

__global__ void zero_kernel(float* __restrict__ p, int n)
{
    int i = blockIdx.x * blockDim.x + threadIdx.x;
    if (i < n) p[i] = 0.f;
}

// u[i] = log_mu[i] - logsumexp_j( Z[i,j] + v[j] )
__global__ __launch_bounds__(256)
void sink_row(const float* __restrict__ Z, const float* __restrict__ vv,
              float* __restrict__ uu, int R, float norm, float logN)
{
    __shared__ float red[256];
    int i = blockIdx.x;
    const float* row = Z + (size_t)i * R;
    float mx = -1e30f;
    for (int j = threadIdx.x; j < R; j += 256) mx = fmaxf(mx, row[j] + vv[j]);
    red[threadIdx.x] = mx; __syncthreads();
    for (int o = 128; o > 0; o >>= 1) {
        if (threadIdx.x < o) red[threadIdx.x] = fmaxf(red[threadIdx.x], red[threadIdx.x + o]);
        __syncthreads();
    }
    mx = red[0]; __syncthreads();
    float s = 0.f;
    for (int j = threadIdx.x; j < R; j += 256) s += __expf(row[j] + vv[j] - mx);
    red[threadIdx.x] = s; __syncthreads();
    for (int o = 128; o > 0; o >>= 1) {
        if (threadIdx.x < o) red[threadIdx.x] += red[threadIdx.x + o];
        __syncthreads();
    }
    if (threadIdx.x == 0) {
        float lse = mx + __logf(red[0]);
        float lmu = (i < R - 1) ? norm : (logN + norm);
        uu[i] = lmu - lse;
    }
}

// v[j] = log_nu[j] - logsumexp_i( Z[i,j] + u[i] )
__global__ __launch_bounds__(256)
void sink_col(const float* __restrict__ Z, const float* __restrict__ uu,
              float* __restrict__ vv, int R, float norm, float logN)
{
    __shared__ float red[256];
    int j = blockIdx.x;
    float mx = -1e30f;
    for (int i = threadIdx.x; i < R; i += 256) mx = fmaxf(mx, Z[(size_t)i * R + j] + uu[i]);
    red[threadIdx.x] = mx; __syncthreads();
    for (int o = 128; o > 0; o >>= 1) {
        if (threadIdx.x < o) red[threadIdx.x] = fmaxf(red[threadIdx.x], red[threadIdx.x + o]);
        __syncthreads();
    }
    mx = red[0]; __syncthreads();
    float s = 0.f;
    for (int i = threadIdx.x; i < R; i += 256) s += __expf(Z[(size_t)i * R + j] + uu[i] - mx);
    red[threadIdx.x] = s; __syncthreads();
    for (int o = 128; o > 0; o >>= 1) {
        if (threadIdx.x < o) red[threadIdx.x] += red[threadIdx.x + o];
        __syncthreads();
    }
    if (threadIdx.x == 0) {
        float lse = mx + __logf(red[0]);
        float lnu = (j < R - 1) ? norm : (logN + norm);
        vv[j] = lnu - lse;
    }
}

// out = Z + u[:,None] + v[None,:] - norm
__global__ void final_out(const float* __restrict__ Z, const float* __restrict__ uu,
                          const float* __restrict__ vv, float* __restrict__ out,
                          int R, float norm)
{
    int j = blockIdx.x * blockDim.x + threadIdx.x;
    int i = blockIdx.y;
    if (j >= R) return;
    out[i * R + j] = Z[i * R + j] + uu[i] + vv[j] - norm;
}

static void launch_gemm(hipStream_t s, const float* A, const float* B, float* C,
                        int M, int N, int K, int as_m, int as_k, int bs_k, int bs_n,
                        int cs_m, int cs_n, const float* bias, const float* res,
                        float scale, int relu)
{
    dim3 g((N + BN - 1) / BN, (M + BM - 1) / BM);
    gemm_wmma<<<g, 256, 0, s>>>(A, B, C, M, N, K, as_m, as_k, bs_k, bs_n,
                                cs_m, cs_n, bias, res, scale, relu);
}

extern "C" void kernel_launch(void* const* d_in, const int* in_sizes, int n_in,
                              void* d_out, int out_size, void* d_ws, size_t ws_size,
                              hipStream_t stream)
{
    (void)in_sizes; (void)n_in; (void)out_size; (void)ws_size;
    const float* kpts0 = (const float*)d_in[0];
    const float* kpts1 = (const float*)d_in[1];
    const float* desc0 = (const float*)d_in[2];
    const float* desc1 = (const float*)d_in[3];
    const float* sc0   = (const float*)d_in[4];
    const float* sc1   = (const float*)d_in[5];
    const float* kw[4] = {(const float*)d_in[6], (const float*)d_in[8],
                          (const float*)d_in[10], (const float*)d_in[12]};
    const float* kb[4] = {(const float*)d_in[7], (const float*)d_in[9],
                          (const float*)d_in[11], (const float*)d_in[13]};
    const float* projW  = (const float*)d_in[14];
    const float* projB  = (const float*)d_in[15];
    const float* mergeW = (const float*)d_in[16];
    const float* mergeB = (const float*)d_in[17];
    const float* W0 = (const float*)d_in[18];
    const float* B0 = (const float*)d_in[19];
    const float* W1 = (const float*)d_in[20];
    const float* B1 = (const float*)d_in[21];
    const float* finW  = (const float*)d_in[22];
    const float* finB  = (const float*)d_in[23];
    const float* alpha = (const float*)d_in[24];

    const int Npts = 1024, D = 256, L = 18, R = 1025, ITERS = 100;

    // workspace layout (floats), ~24 MB total
    float* ws   = (float*)d_ws;
    float* cat0 = ws;                       // [512 x 1024]  rows 0-255 = d0, 256-511 = msg0
    float* cat1 = cat0 + 512 * 1024;
    float* qkvA = cat1 + 512 * 1024;        // [768 x 1024]
    float* qkvB = qkvA + 768 * 1024;
    float* Sbuf = qkvB + 768 * 1024;        // [1024 x 1024]
    float* obuf = Sbuf + 1024 * 1024;       // [256 x 1024]
    float* hbuf = obuf + 256 * 1024;        // [512 x 1024]
    float* md0  = hbuf + 512 * 1024;
    float* md1  = md0 + 256 * 1024;
    float* Zb   = md1 + 256 * 1024;         // [1025 x 1025]
    float* ub   = Zb + R * R;               // [1056]
    float* vb   = ub + 1056;                // [1056]
    float* d0 = cat0;  float* msg0 = cat0 + 256 * 1024;
    float* d1 = cat1;  float* msg1 = cat1 + 256 * 1024;

    // ---- keypoint encoder: d = desc + MLP([kpts^T; scores]) ----
    for (int img = 0; img < 2; ++img) {
        const float* kp = img ? kpts1 : kpts0;
        const float* sc = img ? sc1 : sc0;
        const float* de = img ? desc1 : desc0;
        float* dd = img ? d1 : d0;
        float* kin = obuf;
        float* t0 = hbuf;
        float* t1 = hbuf + 32 * 1024;
        float* t2 = hbuf + 96 * 1024;
        build_kin<<<dim3(4), 256, 0, stream>>>(kp, sc, kin, Npts);
        conv_naive<<<dim3(4, 32), 256, 0, stream>>>(kin, kw[0], kb[0], nullptr, t0, 32, 4, Npts, 0);
        inorm_kernel<<<32, 256, 0, stream>>>(t0, Npts, 1);
        conv_naive<<<dim3(4, 64), 256, 0, stream>>>(t0, kw[1], kb[1], nullptr, t1, 64, 32, Npts, 0);
        inorm_kernel<<<64, 256, 0, stream>>>(t1, Npts, 1);
        conv_naive<<<dim3(4, 128), 256, 0, stream>>>(t1, kw[2], kb[2], nullptr, t2, 128, 64, Npts, 0);
        inorm_kernel<<<128, 256, 0, stream>>>(t2, Npts, 1);
        conv_naive<<<dim3(4, 256), 256, 0, stream>>>(t2, kw[3], kb[3], de, dd, 256, 128, Npts, 0);
    }

    // ---- GNN: 18 layers, all GEMMs via WMMA ----
    for (int i = 0; i < L; ++i) {
        const float* pW = projW  + (size_t)i * 3 * D * D;   // [768 x 256] fused q/k/v
        const float* pB = projB  + (size_t)i * 3 * D;
        const float* mW = mergeW + (size_t)i * D * D;
        const float* mB = mergeB + (size_t)i * D;
        const float* w0 = W0 + (size_t)i * 2 * D * 2 * D;
        const float* b0 = B0 + (size_t)i * 2 * D;
        const float* w1 = W1 + (size_t)i * D * 2 * D;
        const float* b1 = B1 + (size_t)i * D;
        const int cross = i & 1;

        launch_gemm(stream, pW, d0, qkvA, 768, Npts, D, D, 1, Npts, 1, Npts, 1, pB, nullptr, 1.f, 0);
        launch_gemm(stream, pW, d1, qkvB, 768, Npts, D, D, 1, Npts, 1, Npts, 1, pB, nullptr, 1.f, 0);

        for (int img = 0; img < 2; ++img) {
            float* q  = img ? qkvB : qkvA;                              // q from x
            float* sq = img ? (cross ? qkvA : qkvB)
                            : (cross ? qkvB : qkvA);                    // k,v from src
            float* kk = sq + 256 * 1024;
            float* vv = sq + 512 * 1024;
            float* msg = img ? msg1 : msg0;
            // channel c = d*4 + h (reshape (256,n)->(64,4,n)); head view stride 4096
            for (int h = 0; h < 4; ++h) {
                // S[n,m] = sum_d Q_h(d,n) K_h(d,m) / sqrt(64)
                launch_gemm(stream, q + h * 1024, kk + h * 1024, Sbuf,
                            Npts, Npts, 64, 1, 4096, 4096, 1, Npts, 1,
                            nullptr, nullptr, 0.125f, 0);
                softmax_rows<<<Npts, 256, 0, stream>>>(Sbuf, Npts);
                // O_h(n,d) = sum_m P[n,m] V_h(d,m)
                launch_gemm(stream, Sbuf, vv + h * 1024, obuf + h * 1024,
                            Npts, 64, Npts, Npts, 1, 1, 4096, 1, 4096,
                            nullptr, nullptr, 1.f, 0);
            }
            launch_gemm(stream, mW, obuf, msg, D, Npts, D, D, 1, Npts, 1, Npts, 1,
                        mB, nullptr, 1.f, 0);
        }

        for (int img = 0; img < 2; ++img) {
            float* cat = img ? cat1 : cat0;
            float* dd  = img ? d1 : d0;
            launch_gemm(stream, w0, cat, hbuf, 512, Npts, 512, 512, 1, Npts, 1, Npts, 1,
                        b0, nullptr, 1.f, 0);
            inorm_kernel<<<512, 256, 0, stream>>>(hbuf, Npts, 1);
            launch_gemm(stream, w1, hbuf, dd, D, Npts, 512, 512, 1, Npts, 1, Npts, 1,
                        b1, dd, 1.f, 0);
        }
    }

    // ---- final projection + matching scores ----
    launch_gemm(stream, finW, d0, md0, D, Npts, D, D, 1, Npts, 1, Npts, 1, finB, nullptr, 1.f, 0);
    launch_gemm(stream, finW, d1, md1, D, Npts, D, D, 1, Npts, 1, Npts, 1, finB, nullptr, 1.f, 0);
    launch_gemm(stream, md0, md1, Sbuf, Npts, Npts, D, 1, Npts, Npts, 1, Npts, 1,
                nullptr, nullptr, 0.0625f, 0);

    // ---- Sinkhorn in log space ----
    build_couplings<<<dim3((R + 255) / 256, R), 256, 0, stream>>>(Sbuf, alpha, Zb, R);
    zero_kernel<<<dim3((2 * 1056 + 255) / 256), 256, 0, stream>>>(ub, 2 * 1056);
    const float norm = -logf(2048.f);
    const float logN = logf(1024.f);
    for (int it = 0; it < ITERS; ++it) {
        sink_row<<<R, 256, 0, stream>>>(Zb, vb, ub, R, norm, logN);
        sink_col<<<R, 256, 0, stream>>>(Zb, ub, vb, R, norm, logN);
    }
    final_out<<<dim3((R + 255) / 256, R), 256, 0, stream>>>(Zb, ub, vb, (float*)d_out, R, norm);
}
